// MultiHeadAttention_20968030339462
// MI455X (gfx1250) — compile-verified, matrix operand order checked by reading, not run on hardware
//
#include <hip/hip_runtime.h>

// ---------------------------------------------------------------------------
// Shapes (fixed by the reference)
// ---------------------------------------------------------------------------
#define NTOK 1024            // IH*IW
#define HEADS 8
#define DIMH 32
#define INNER 256            // HEADS*DIMH
#define INP 256
#define OUTD 256
#define BATCH 16

typedef __attribute__((ext_vector_type(16))) __bf16 v16bf;
typedef __attribute__((ext_vector_type(8)))  float  v8f;
typedef int v4i __attribute__((vector_size(16)));

union FragU { v16bf bf; uint4 q[2]; };

#if defined(__has_builtin)
#if __has_builtin(__builtin_amdgcn_global_load_async_to_lds_b128)
#define USE_ASYNC_LDS 1
#endif
#endif

#ifdef USE_ASYNC_LDS
// CDNA5 async global->LDS copy, 16B per lane, tracked by ASYNCcnt.
__device__ __forceinline__ void async_load_b128(const void* g, void* l) {
    __builtin_amdgcn_global_load_async_to_lds_b128(
        (__attribute__((address_space(1))) v4i*)(void*)g,
        (__attribute__((address_space(3))) v4i*)l,
        0, 0);
}
#endif

__device__ __forceinline__ void wait_async0() {
#if defined(__has_builtin) && __has_builtin(__builtin_amdgcn_s_wait_asynccnt)
    __builtin_amdgcn_s_wait_asynccnt(0);
#else
    asm volatile("s_wait_asynccnt 0x0" ::: "memory");
#endif
}

// float -> bf16 round-to-nearest-even
__device__ __forceinline__ unsigned short f2bf(float f) {
    unsigned int u = __builtin_bit_cast(unsigned int, f);
    u += 0x7fffu + ((u >> 16) & 1u);
    return (unsigned short)(u >> 16);
}

// Load a 16x32 bf16 A/B operand fragment (ISA 7.12.2 16-bit layout).
// base points at the first row of the 16-row operand tile (row-major, u16).
// Lane l: row = l&15, half = l>>4 selects K {0..7,16..23} or {8..15,24..31}.
__device__ __forceinline__ v16bf load_frag(const unsigned short* base,
                                           int idx, int rowStride, int half) {
    FragU f;
    const unsigned short* p = base + idx * rowStride + half * 8;
    f.q[0] = *(const uint4*)(p);
    f.q[1] = *(const uint4*)(p + 16);
    return f.bf;
}

__device__ __forceinline__ v8f wmma_bf16(v16bf a, v16bf b, v8f c) {
    return __builtin_amdgcn_wmma_f32_16x16x32_bf16(
        /*neg_a=*/false, a, /*neg_b=*/false, b,
        /*c_mod=*/(short)0, c, /*reuse_a=*/false, /*reuse_b=*/false);
}

// ---------------------------------------------------------------------------
// Kernel 1: bias[h][i][j] = rel_table[rel_index[j*N+i]*H + h]
// 32 MB fp32 table -> resident in the 192 MB L2 for the attention pass.
// ---------------------------------------------------------------------------
__global__ __launch_bounds__(256) void bias_build(
        const float* __restrict__ rel_table,
        const int*   __restrict__ rel_index,
        float*       __restrict__ Bias) {
    int t = blockIdx.x * 256 + threadIdx.x;       // 0 .. N*N-1
    int i = t >> 10;
    int j = t & 1023;
    int ridx = rel_index[j * NTOK + i];
    const float* rt = rel_table + (size_t)ridx * HEADS;
#pragma unroll
    for (int h = 0; h < HEADS; ++h)
        Bias[(size_t)h * NTOK * NTOK + (size_t)i * NTOK + j] = rt[h];
}

// ---------------------------------------------------------------------------
// Kernel 2: qkv = x @ w_qkv ([16384,256] @ [256,768]), bf16 WMMA, f32 accum.
// Block tile 128x128, 8 waves in 4x2 grid, each wave 32x64 (8 WMMAs/k-step).
// Epilogue: one block-uniform branch selects q/k/v; per 16x16 tile a single
// base address + 8 stores with immediate offsets.
// ---------------------------------------------------------------------------
__global__ __launch_bounds__(256) void qkv_gemm(
        const float* __restrict__ X, const float* __restrict__ W,
        unsigned short* __restrict__ Qb, unsigned short* __restrict__ Kb,
        unsigned short* __restrict__ Vb) {
    __shared__ __align__(16) unsigned short At[128 * 32];   // [m][k]
    __shared__ __align__(16) unsigned short Bt[128 * 32];   // [n][k] (transposed)

    const int tid  = threadIdx.x;
    const int lane = tid & 31;
    const int wave = tid >> 5;
    const int half = lane >> 4;
    const int idx  = lane & 15;
    const int wm   = wave >> 1;          // 0..3
    const int wn   = wave & 1;           // 0..1
    const int m0   = blockIdx.x * 128;
    const int n0   = blockIdx.y * 128;

    v8f zero = {};
    v8f acc[2][4];
#pragma unroll
    for (int i = 0; i < 2; ++i)
#pragma unroll
        for (int j = 0; j < 4; ++j) acc[i][j] = zero;

    for (int kk = 0; kk < INP; kk += 32) {
        __syncthreads();
        {   // stage A 128x32 f32 -> bf16 (row-major)
            int row = tid >> 1;
            int kb  = (tid & 1) * 16;
            const float* src = X + (size_t)(m0 + row) * INP + kk + kb;
            unsigned short* dst = At + row * 32 + kb;
#pragma unroll
            for (int u = 0; u < 16; ++u) dst[u] = f2bf(src[u]);
        }
        {   // stage B 32(k)x128(n) f32 -> bf16, transposed into [n][k]
            int krow = tid >> 3;
            int nb   = (tid & 7) * 16;
            const float* src = W + (size_t)(kk + krow) * (3 * INNER) + n0 + nb;
#pragma unroll
            for (int u = 0; u < 16; ++u) Bt[(nb + u) * 32 + krow] = f2bf(src[u]);
        }
        __syncthreads();

        v16bf a0 = load_frag(At + (wm * 32 + 0)  * 32, idx, 32, half);
        v16bf a1 = load_frag(At + (wm * 32 + 16) * 32, idx, 32, half);
        v16bf bf[4];
#pragma unroll
        for (int j = 0; j < 4; ++j)
            bf[j] = load_frag(Bt + (wn * 64 + j * 16) * 32, idx, 32, half);
#pragma unroll
        for (int j = 0; j < 4; ++j) {
            acc[0][j] = wmma_bf16(a0, bf[j], acc[0][j]);
            acc[1][j] = wmma_bf16(a1, bf[j], acc[1][j]);
        }
    }

    // ---- Epilogue ---------------------------------------------------------
    // Block-uniform: part (q/k/v) and batch index b.
    const int part  = n0 >> 8;                    // 0=q 1=k 2=v
    const int cbase = n0 & 255;
    const int b     = m0 >> 10;                   // 128-row tiles never cross b
    const int rb    = (m0 & 1023) + wm * 32 + 8 * half;  // + i*16 + r

    if (part < 2) {   // Q or K: [B,H,N,D] row-major; r steps irow (+32 elems)
        unsigned short* __restrict__ QK = (part == 0) ? Qb : Kb;
#pragma unroll
        for (int i = 0; i < 2; ++i)
#pragma unroll
        for (int j = 0; j < 4; ++j) {
            int c    = cbase + wn * 64 + j * 16;        // tile col base
            int hh   = c >> 5;                          // uniform in tile
            int d    = (c & 31) + idx;
            int irow = rb + i * 16;
            unsigned short* p =
                QK + ((size_t)(b * HEADS + hh) * NTOK + irow) * DIMH + d;
#pragma unroll
            for (int r = 0; r < 8; ++r)
                p[r * DIMH] = f2bf(acc[i][j][r]);
        }
    } else {          // V transposed: [B,H,D,N]; r steps irow (+1 elem)
#pragma unroll
        for (int i = 0; i < 2; ++i)
#pragma unroll
        for (int j = 0; j < 4; ++j) {
            int c    = cbase + wn * 64 + j * 16;
            int hh   = c >> 5;
            int d    = (c & 31) + idx;
            int irow = rb + i * 16;
            unsigned short* p =
                Vb + ((size_t)(b * HEADS + hh) * DIMH + d) * NTOK + irow;
#pragma unroll
            for (int r = 0; r < 8; ++r)
                p[r] = f2bf(acc[i][j][r]);
        }
    }
}

// ---------------------------------------------------------------------------
// Kernel 3: flash attention. Grid (N/128, H, B); 8 waves, 16 query rows each.
// Per 32-key step/wave: 2 WMMA (S = Q K^T) + online softmax + 2 WMMA (O += PV)
// K/V tiles staged with CDNA5 async global->LDS loads (ASYNCcnt) when present.
// ---------------------------------------------------------------------------
__global__ __launch_bounds__(256) void attn_kernel(
        const unsigned short* __restrict__ Qb,
        const unsigned short* __restrict__ Kb,
        const unsigned short* __restrict__ Vb,
        const float* __restrict__ Bias,
        float* __restrict__ AttO) {
    __shared__ __align__(16) unsigned short Kt[32 * 32];      // [j][d]
    __shared__ __align__(16) unsigned short Vt[32 * 32];      // [d][j]
    __shared__ __align__(16) unsigned short Pb[8 * 16 * 32];  // per-wave P

    const int tid  = threadIdx.x;
    const int lane = tid & 31;
    const int wave = tid >> 5;
    const int half = lane >> 4;
    const int idx  = lane & 15;
    const int h    = blockIdx.y;
    const int b    = blockIdx.z;
    const size_t bh = (size_t)b * HEADS + h;
    const int i0   = blockIdx.x * 128 + wave * 16;   // wave's query-row base

    const float L2E = 1.44269504088896340736f;

    // Resident Q fragment (16 rows x 32 dims, bf16)
    v16bf qf = load_frag(Qb + (bh * NTOK + i0) * DIMH, idx, DIMH, half);

    const float* biasH = Bias + (size_t)h * NTOK * NTOK;

    v8f zero = {};
    v8f acc0 = zero, acc1 = zero;           // O tiles: d 0..15 and 16..31
    float mrow[8], lrow[8];
#pragma unroll
    for (int r = 0; r < 8; ++r) { mrow[r] = -3.0e38f; lrow[r] = 0.0f; }

    for (int j0 = 0; j0 < NTOK; j0 += 32) {
        __syncthreads();
#ifdef USE_ASYNC_LDS
        {   // async staging: 128 lanes x 16B per tile (ASYNCcnt tracked)
            if (tid < 128) {               // K tile [32 j][32 d]
                int row = tid >> 2;
                int cb  = (tid & 3) * 8;   // bf16 elements (16B chunks)
                async_load_b128(Kb + (bh * NTOK + j0 + row) * DIMH + cb,
                                Kt + row * 32 + cb);
            } else {                       // V tile [32 d][32 j]
                int t   = tid - 128;
                int row = t >> 2;
                int cb  = (t & 3) * 8;
                async_load_b128(Vb + (bh * DIMH + row) * NTOK + j0 + cb,
                                Vt + row * 32 + cb);
            }
            wait_async0();
        }
#else
        {   // fallback synchronous staging
            int row = tid >> 3;
            int cb  = (tid & 7) * 4;
            *(uint2*)(Kt + row * 32 + cb) =
                *(const uint2*)(Kb + (bh * NTOK + j0 + row) * DIMH + cb);
            *(uint2*)(Vt + row * 32 + cb) =
                *(const uint2*)(Vb + (bh * DIMH + row) * NTOK + j0 + cb);
        }
#endif
        __syncthreads();

        v16bf k0 = load_frag(Kt,           idx, 32, half);   // keys j0+0..15
        v16bf k1 = load_frag(Kt + 16 * 32, idx, 32, half);   // keys j0+16..31
        v8f s0 = wmma_bf16(qf, k0, zero);
        v8f s1 = wmma_bf16(qf, k1, zero);

        float p0[8], p1[8], corr[8];
#pragma unroll
        for (int r = 0; r < 8; ++r) {
            int gi = i0 + r + 8 * half;
            const float* bp = biasH + (size_t)gi * NTOK + j0 + idx;
            float t0 = s0[r] + bp[0];
            float t1 = s1[r] + bp[16];
            float mx = fmaxf(t0, t1);
            mx = fmaxf(mx, __shfl_xor(mx, 1, 32));
            mx = fmaxf(mx, __shfl_xor(mx, 2, 32));
            mx = fmaxf(mx, __shfl_xor(mx, 4, 32));
            mx = fmaxf(mx, __shfl_xor(mx, 8, 32));   // stays within 16-lane half
            float mn = fmaxf(mrow[r], mx);
            float c  = exp2f((mrow[r] - mn) * L2E);
            p0[r] = exp2f((t0 - mn) * L2E);
            p1[r] = exp2f((t1 - mn) * L2E);
            float rs = p0[r] + p1[r];
            rs += __shfl_xor(rs, 1, 32);
            rs += __shfl_xor(rs, 2, 32);
            rs += __shfl_xor(rs, 4, 32);
            rs += __shfl_xor(rs, 8, 32);
            lrow[r] = lrow[r] * c + rs;
            mrow[r] = mn;
            corr[r] = c;
        }
#pragma unroll
        for (int r = 0; r < 8; ++r) { acc0[r] *= corr[r]; acc1[r] *= corr[r]; }

        // P: D-layout -> A-layout via per-wave LDS bounce (bf16)
        unsigned short* pw = Pb + wave * 16 * 32;
#pragma unroll
        for (int r = 0; r < 8; ++r) {
            int row = r + 8 * half;
            pw[row * 32 + idx]      = f2bf(p0[r]);
            pw[row * 32 + 16 + idx] = f2bf(p1[r]);
        }
        asm volatile("s_wait_dscnt 0" ::: "memory");  // wave-local LDS RAW

        v16bf pf = load_frag(pw,           idx, 32, half);
        v16bf v0 = load_frag(Vt,           idx, 32, half);   // d 0..15
        v16bf v1 = load_frag(Vt + 16 * 32, idx, 32, half);   // d 16..31
        acc0 = wmma_bf16(pf, v0, acc0);
        acc1 = wmma_bf16(pf, v1, acc1);
    }

    // Normalize and write attention output [B, N, H*D] fp32
#pragma unroll
    for (int r = 0; r < 8; ++r) {
        int gi = i0 + r + 8 * half;
        float inv = 1.0f / lrow[r];
        size_t o = ((size_t)b * NTOK + gi) * INNER + h * DIMH;
        AttO[o + idx]      = acc0[r] * inv;
        AttO[o + 16 + idx] = acc1[r] * inv;
    }
}

// ---------------------------------------------------------------------------
// Kernel 4: out = attnO @ w_out + b_out ([16384,256] @ [256,256]) -> fp32
// Same 128x128 block / 32x64 wave tiling as kernel 2.
// ---------------------------------------------------------------------------
__global__ __launch_bounds__(256) void out_gemm(
        const float* __restrict__ X, const float* __restrict__ W,
        const float* __restrict__ bias, float* __restrict__ Out) {
    __shared__ __align__(16) unsigned short At[128 * 32];
    __shared__ __align__(16) unsigned short Bt[128 * 32];

    const int tid  = threadIdx.x;
    const int lane = tid & 31;
    const int wave = tid >> 5;
    const int half = lane >> 4;
    const int idx  = lane & 15;
    const int wm   = wave >> 1;
    const int wn   = wave & 1;
    const int m0   = blockIdx.x * 128;
    const int n0   = blockIdx.y * 128;

    v8f zero = {};
    v8f acc[2][4];
#pragma unroll
    for (int i = 0; i < 2; ++i)
#pragma unroll
        for (int j = 0; j < 4; ++j) acc[i][j] = zero;

    for (int kk = 0; kk < INNER; kk += 32) {
        __syncthreads();
        {
            int row = tid >> 1;
            int kb  = (tid & 1) * 16;
            const float* src = X + (size_t)(m0 + row) * INNER + kk + kb;
            unsigned short* dst = At + row * 32 + kb;
#pragma unroll
            for (int u = 0; u < 16; ++u) dst[u] = f2bf(src[u]);
        }
        {
            int krow = tid >> 3;
            int nb   = (tid & 7) * 16;
            const float* src = W + (size_t)(kk + krow) * OUTD + n0 + nb;
#pragma unroll
            for (int u = 0; u < 16; ++u) Bt[(nb + u) * 32 + krow] = f2bf(src[u]);
        }
        __syncthreads();

        v16bf a0 = load_frag(At + (wm * 32 + 0)  * 32, idx, 32, half);
        v16bf a1 = load_frag(At + (wm * 32 + 16) * 32, idx, 32, half);
        v16bf bf[4];
#pragma unroll
        for (int j = 0; j < 4; ++j)
            bf[j] = load_frag(Bt + (wn * 64 + j * 16) * 32, idx, 32, half);
#pragma unroll
        for (int j = 0; j < 4; ++j) {
            acc[0][j] = wmma_bf16(a0, bf[j], acc[0][j]);
            acc[1][j] = wmma_bf16(a1, bf[j], acc[1][j]);
        }
    }

#pragma unroll
    for (int i = 0; i < 2; ++i)
#pragma unroll
    for (int j = 0; j < 4; ++j) {
        int grow = m0 + wm * 32 + i * 16 + 8 * half;
        int gcol = n0 + wn * 64 + j * 16 + idx;
        float bb = bias[gcol];
        float* p = Out + (size_t)grow * OUTD + gcol;
#pragma unroll
        for (int r = 0; r < 8; ++r)
            p[(size_t)r * OUTD] = acc[i][j][r] + bb;
    }
}

// ---------------------------------------------------------------------------
// Launch: inputs are {x, w_qkv, rel_table, w_out, b_out, rel_index}
// ---------------------------------------------------------------------------
extern "C" void kernel_launch(void* const* d_in, const int* in_sizes, int n_in,
                              void* d_out, int out_size, void* d_ws, size_t ws_size,
                              hipStream_t stream) {
    const float* x         = (const float*)d_in[0];
    const float* w_qkv     = (const float*)d_in[1];
    const float* rel_table = (const float*)d_in[2];
    const float* w_out     = (const float*)d_in[3];
    const float* b_out     = (const float*)d_in[4];
    const int*   rel_index = (const int*)d_in[5];

    char* ws = (char*)d_ws;
    const size_t MB = 1024 * 1024;
    unsigned short* Qb  = (unsigned short*)(ws);              //  8 MB bf16
    unsigned short* Kb  = (unsigned short*)(ws + 8 * MB);     //  8 MB bf16
    unsigned short* Vb  = (unsigned short*)(ws + 16 * MB);    //  8 MB bf16 (V^T)
    float*          Bia = (float*)(ws + 24 * MB);             // 32 MB fp32
    float*          AtO = (float*)(ws + 56 * MB);             // 16 MB fp32
    float*          Out = (float*)d_out;

    bias_build<<<dim3((NTOK * NTOK) / 256), dim3(256), 0, stream>>>(
        rel_table, rel_index, Bia);

    qkv_gemm<<<dim3((BATCH * NTOK) / 128, (3 * INNER) / 128), dim3(256), 0, stream>>>(
        x, w_qkv, Qb, Kb, Vb);

    attn_kernel<<<dim3(NTOK / 128, HEADS, BATCH), dim3(256), 0, stream>>>(
        Qb, Kb, Vb, Bia, AtO);

    out_gemm<<<dim3((BATCH * NTOK) / 128, OUTD / 128), dim3(256), 0, stream>>>(
        AtO, w_out, b_out, Out);
}
